// MambaNet_26783416058192
// MI455X (gfx1250) — matrix operand order repaired and checked by reading, not compiled, not run on hardware
//
#include <hip/hip_runtime.h>
#include <math.h>

// ---------------------------------------------------------------------------
// MI455X (gfx1250, wave32) implementation of the MambaNet reference.
// All GEMMs -> v_wmma_f32_16x16x32_bf16 (bf16 A/B, f32 accum).
// Big GEMMs: 4-wave workgroup, 64x64 tile, double-buffered LDS B stage fed by
// GLOBAL_LOAD_ASYNC_TO_LDS_B128 (ASYNCcnt), consumed via ds_load_b128.
// Elementwise / scan math stays f32 for accuracy.
// ---------------------------------------------------------------------------

typedef __attribute__((ext_vector_type(16))) __bf16 bf16x16;
typedef __attribute__((ext_vector_type(8)))  __bf16 bf16x8;
typedef __attribute__((ext_vector_type(8)))  float  f32x8;

union BF16Frag { bf16x16 v; bf16x8 h[2]; };

#ifndef __has_builtin
#define __has_builtin(x) 0
#endif
#if __has_builtin(__builtin_amdgcn_global_load_async_to_lds_b128) && \
    __has_builtin(__builtin_amdgcn_s_wait_asynccnt)
#define USE_ASYNC_LDS 1
#else
#define USE_ASYNC_LDS 0
#endif

#if USE_ASYNC_LDS
typedef int v4i __attribute__((vector_size(4 * sizeof(int))));
typedef __attribute__((address_space(1))) v4i* gas1_v4i;
typedef __attribute__((address_space(3))) v4i* las3_v4i;
#endif

// Copy 16 bytes global -> LDS (async on gfx1250).
static __device__ inline void stage16B(const __bf16* g, __bf16* l)
{
#if USE_ASYNC_LDS
    __builtin_amdgcn_global_load_async_to_lds_b128((gas1_v4i)(g), (las3_v4i)(l), 0, 0);
#else
    *(bf16x8*)l = *(const bf16x8*)g;
#endif
}

static __device__ inline void stage_wait()
{
#if USE_ASYNC_LDS
    __builtin_amdgcn_s_wait_asynccnt(0);
#endif
    __syncthreads();
}

// ---------------------------------------------------------------------------
// Simple 1-wave GEMM (used only for shapes not multiple of 64):
//   C[M,N] (+)= A[M,K] * B[K,N],  B transposed: B[k,n] = Bt[n*ldbt + k]
// ---------------------------------------------------------------------------
__global__ void gemm_bf16_wmma(const __bf16* __restrict__ A,
                               const __bf16* __restrict__ Bt,
                               float* __restrict__ C,
                               int K, int lda, int ldbt, int ldc,
                               long long sA, long long sB, long long sC,
                               int accum)
{
    const int tn = blockIdx.x;
    const int tm = blockIdx.y;
    const int bz = blockIdx.z;
    A  += (long long)bz * sA;
    Bt += (long long)bz * sB;
    C  += (long long)bz * sC;

    const int lane = threadIdx.x & 31;
    const int lo = lane & 15;
    const int hi = lane >> 4;

    const __bf16* ar = A  + (long long)(tm * 16 + lo) * lda + hi * 16;
    const __bf16* br = Bt + (long long)(tn * 16 + lo) * ldbt + hi * 16;
    float* cr = C + (long long)(tm * 16 + hi * 8) * ldc + tn * 16 + lo;

    f32x8 acc = {0.f, 0.f, 0.f, 0.f, 0.f, 0.f, 0.f, 0.f};
    if (accum) {
#pragma unroll
        for (int p = 0; p < 8; ++p) acc[p] = cr[(long long)p * ldc];
    }

    for (int k0 = 0; k0 < K; k0 += 32) {
        BF16Frag a, b;
        a.h[0] = *(const bf16x8*)(ar + k0);
        a.h[1] = *(const bf16x8*)(ar + k0 + 8);
        b.h[0] = *(const bf16x8*)(br + k0);
        b.h[1] = *(const bf16x8*)(br + k0 + 8);
        acc = __builtin_amdgcn_wmma_f32_16x16x32_bf16(
            false, a.v, false, b.v, (short)0, acc, false, false);
    }

#pragma unroll
    for (int p = 0; p < 8; ++p) cr[(long long)p * ldc] = acc[p];
}

// ---------------------------------------------------------------------------
// Blocked GEMM: 128 threads = 4 waves. Block tile 64(M) x 64(N).
//  - wave w owns M rows [m0+16w, m0+16w+16), all 4 N sub-tiles (A frag reused)
//  - B tile (64 rows x 32 k) staged in LDS, double buffered, shared by waves
//  - LDS row stride 40 elems (80B): 16B aligned ds_load_b128, conflict-free
// Requires M%64==0, N%64==0, K%32==0.
// ---------------------------------------------------------------------------
#define LB_STRIDE 40
#define LB_BUF    (64 * LB_STRIDE)

__global__ void __launch_bounds__(128)
gemm_bf16_wmma_blk(const __bf16* __restrict__ A,
                   const __bf16* __restrict__ Bt,
                   float* __restrict__ C,
                   int K, int lda, int ldbt, int ldc,
                   long long sA, long long sB, long long sC,
                   int accum)
{
    __shared__ __bf16 lb[2 * LB_BUF];

    const int bz = blockIdx.z;
    A  += (long long)bz * sA;
    Bt += (long long)bz * sB;
    C  += (long long)bz * sC;

    const int tid  = threadIdx.x;
    const int wave = tid >> 5;
    const int lane = tid & 31;
    const int lo = lane & 15, hi = lane >> 4;

    const int m0 = blockIdx.y * 64;
    const int n0 = blockIdx.x * 64;

    const __bf16* ar = A + (long long)(m0 + wave * 16 + lo) * lda + hi * 16;

    // cooperative B staging: 256 16B chunks; thread handles chunks tid, tid+128
    const int c0 = tid, c1 = tid + 128;
    const int r0 = c0 >> 2, kp0 = (c0 & 3) * 8;
    const int r1 = c1 >> 2, kp1 = (c1 & 3) * 8;
    const __bf16* bp0 = Bt + (long long)(n0 + r0) * ldbt + kp0;
    const __bf16* bp1 = Bt + (long long)(n0 + r1) * ldbt + kp1;
    __bf16* lw0 = &lb[r0 * LB_STRIDE + kp0];
    __bf16* lw1 = &lb[r1 * LB_STRIDE + kp1];

    f32x8 acc[4];
    float* cr[4];
#pragma unroll
    for (int t = 0; t < 4; ++t) {
        cr[t] = C + (long long)(m0 + wave * 16 + hi * 8) * ldc + n0 + t * 16 + lo;
        acc[t] = (f32x8){0.f, 0.f, 0.f, 0.f, 0.f, 0.f, 0.f, 0.f};
        if (accum) {
#pragma unroll
            for (int p = 0; p < 8; ++p) acc[t][p] = cr[t][(long long)p * ldc];
        }
    }

    // prologue: stage k-block 0 into buffer 0
    stage16B(bp0, lw0);
    stage16B(bp1, lw1);

    int buf = 0;
    for (int k0 = 0; k0 < K; k0 += 32) {
        stage_wait();                        // current buffer ready for all waves
        if (k0 + 32 < K) {                   // issue next stage while computing
            stage16B(bp0 + k0 + 32, lw0 + (buf ^ 1) * LB_BUF);
            stage16B(bp1 + k0 + 32, lw1 + (buf ^ 1) * LB_BUF);
        }
        __builtin_prefetch((const void*)(ar + k0 + 128), 0, 1);

        BF16Frag a;
        a.h[0] = *(const bf16x8*)(ar + k0);
        a.h[1] = *(const bf16x8*)(ar + k0 + 8);
        const __bf16* lbase = &lb[buf * LB_BUF + lo * LB_STRIDE + hi * 16];
#pragma unroll
        for (int t = 0; t < 4; ++t) {
            BF16Frag b;
            const __bf16* lp = lbase + t * 16 * LB_STRIDE;
            b.h[0] = *(const bf16x8*)(lp);
            b.h[1] = *(const bf16x8*)(lp + 8);
            acc[t] = __builtin_amdgcn_wmma_f32_16x16x32_bf16(
                false, a.v, false, b.v, (short)0, acc[t], false, false);
        }
        buf ^= 1;
    }

#pragma unroll
    for (int t = 0; t < 4; ++t)
#pragma unroll
        for (int p = 0; p < 8; ++p) cr[t][(long long)p * ldc] = acc[t][p];
}

// ---------------------------------------------------------------------------
// LayerNorm over last dim (d), one block per row.
// ---------------------------------------------------------------------------
__global__ void layernorm_k(const float* __restrict__ x, const float* __restrict__ g,
                            const float* __restrict__ b, float* __restrict__ o, int d)
{
    const long long row = blockIdx.x;
    const float* xr = x + row * d;
    float* orow = o + row * d;
    __shared__ float red[256];

    float s = 0.f;
    for (int i = threadIdx.x; i < d; i += blockDim.x) s += xr[i];
    red[threadIdx.x] = s;
    __syncthreads();
    for (int off = blockDim.x >> 1; off > 0; off >>= 1) {
        if ((int)threadIdx.x < off) red[threadIdx.x] += red[threadIdx.x + off];
        __syncthreads();
    }
    const float mean = red[0] / d;
    __syncthreads();

    float v = 0.f;
    for (int i = threadIdx.x; i < d; i += blockDim.x) {
        float t = xr[i] - mean;
        v += t * t;
    }
    red[threadIdx.x] = v;
    __syncthreads();
    for (int off = blockDim.x >> 1; off > 0; off >>= 1) {
        if ((int)threadIdx.x < off) red[threadIdx.x] += red[threadIdx.x + off];
        __syncthreads();
    }
    const float rstd = rsqrtf(red[0] / d + 1e-5f);

    for (int i = threadIdx.x; i < d; i += blockDim.x)
        orow[i] = (xr[i] - mean) * rstd * g[i] + b[i];
}

// ---------------------------------------------------------------------------
// Transpose+pad+cvt for the dense seq-conv: per batch, Xt has (d+2) rows of
// length L:  Xt[b][r][i] = (r==0 || r==d+1) ? 0 : X[b][i][r-1]  (bf16)
// Then conv tap t is a pure GEMM with Bt base = Xt + t*L.
// ---------------------------------------------------------------------------
__global__ void transpose_pad_bf16(const float* __restrict__ X, __bf16* __restrict__ Xt,
                                   int L, int d)
{
    const int total = (d + 2) * L;
    int idx = blockIdx.x * blockDim.x + threadIdx.x;
    if (idx >= total) return;
    const int bz = blockIdx.y;
    const int r = idx / L;
    const int i = idx - r * L;
    float v = 0.f;
    if (r != 0 && r != d + 1)
        v = X[((long long)bz * L + i) * d + (r - 1)];
    Xt[(long long)bz * total + idx] = (__bf16)v;
}

// Pack conv weight (O,I,3) -> Wp[tap][O][I] bf16 (contiguous-in-I rows).
__global__ void pack_convw(const float* __restrict__ W, __bf16* __restrict__ Wp, int L)
{
    long long idx = (long long)blockIdx.x * blockDim.x + threadIdx.x;
    const long long total = 3LL * L * L;
    if (idx >= total) return;
    long long tap = idx / ((long long)L * L);
    long long rem = idx - tap * L * L;
    long long o = rem / L, i = rem - o * L;
    Wp[idx] = (__bf16)W[(o * L + i) * 3 + tap];
}

__global__ void cvt_bf16(const float* __restrict__ s, __bf16* __restrict__ d, long long n)
{
    long long idx = (long long)blockIdx.x * blockDim.x + threadIdx.x;
    if (idx < n) d[idx] = (__bf16)s[idx];
}

// Pack with zero row padding: dst is (padRows x cols), src is (rows x cols).
__global__ void pack_pad_bf16(const float* __restrict__ s, __bf16* __restrict__ d,
                              int rows, int cols, int padRows)
{
    int idx = blockIdx.x * blockDim.x + threadIdx.x;
    int total = padRows * cols;
    if (idx >= total) return;
    int r = idx / cols;
    d[idx] = (r < rows) ? (__bf16)s[idx] : (__bf16)0.f;
}

// bias (indexed by sequence channel l) + exact GeLU
__global__ void bias_gelu(const float* __restrict__ c, const float* __restrict__ bias,
                          float* __restrict__ o, int L, int d, long long total)
{
    long long idx = (long long)blockIdx.x * blockDim.x + threadIdx.x;
    if (idx >= total) return;
    int l = (int)((idx / d) % L);
    float v = c[idx] + bias[l];
    o[idx] = 0.5f * v * (1.f + erff(v * 0.70710678118654752f));
}

// Depthwise causal conv (k=4, left pad 3) + SiLU; writes f32 and bf16.
// u[b,t,c] = xz[(b*L+t)*ldxz + c]
__global__ void dwconv_silu(const float* __restrict__ xz, const float* __restrict__ cw,
                            const float* __restrict__ cb, float* __restrict__ u,
                            __bf16* __restrict__ ub, int L, int nch, int ldxz,
                            long long total)
{
    long long idx = (long long)blockIdx.x * blockDim.x + threadIdx.x;
    if (idx >= total) return;
    int c = (int)(idx % nch);
    long long bl = idx / nch;
    int l = (int)(bl % L);
    long long b = bl / L;
    float acc = cb[c];
#pragma unroll
    for (int j = 0; j < 4; ++j) {
        int t = l + j - 3;
        if (t >= 0)
            acc += cw[c * 4 + j] * xz[((long long)(b * L + t)) * ldxz + c];
    }
    float r = acc / (1.f + expf(-acc));
    u[idx] = r;
    ub[idx] = (__bf16)r;
}

// In-place softplus with per-channel bias.
__global__ void softplus_bias(float* __restrict__ dt, const float* __restrict__ db,
                              long long total, int nch)
{
    long long idx = (long long)blockIdx.x * blockDim.x + threadIdx.x;
    if (idx >= total) return;
    float v = dt[idx] + db[idx % nch];
    dt[idx] = (v > 20.f) ? v : log1pf(expf(v));
}

__global__ void negexp(const float* __restrict__ a, float* __restrict__ o, int n)
{
    int idx = blockIdx.x * blockDim.x + threadIdx.x;
    if (idx < n) o[idx] = -expf(a[idx]);
}

// ---------------------------------------------------------------------------
// Selective scan. wave32: 16 lanes = the 16 SSM states of one channel,
// 2 channels per wave, 8 waves (16 channels) per block.
// grid = (nch/16, batch), block = 256.
// ---------------------------------------------------------------------------
__global__ void mamba_scan(const float* __restrict__ dt, const float* __restrict__ u,
                           const float* __restrict__ dbc, int ld_dbc, int offB, int offC,
                           const float* __restrict__ z, int ldz,
                           const float* __restrict__ Aneg, const float* __restrict__ Dv,
                           __bf16* __restrict__ y, int L, int nch)
{
    const int lane = threadIdx.x & 31;
    const int s = lane & 15;
    const int c = blockIdx.x * 16 + (threadIdx.x >> 4);
    const long long b = blockIdx.y;

    const float a = Aneg[c * 16 + s];
    const float dskip = Dv[c];
    float h = 0.f;

    for (int l = 0; l < L; ++l) {
        const long long rb = b * L + l;
        const float dt_t = dt[rb * nch + c];
        const float u_t  = u[rb * nch + c];
        const float B_t  = dbc[rb * ld_dbc + offB + s];
        const float C_t  = dbc[rb * ld_dbc + offC + s];
        h = expf(dt_t * a) * h + (dt_t * u_t) * B_t;
        float yp = h * C_t;
        yp += __shfl_xor(yp, 8, 32);
        yp += __shfl_xor(yp, 4, 32);
        yp += __shfl_xor(yp, 2, 32);
        yp += __shfl_xor(yp, 1, 32);
        if (s == 0) {
            float zt = z[rb * ldz + c];
            float yv = (yp + u_t * dskip) * (zt / (1.f + expf(-zt)));
            y[rb * nch + c] = (__bf16)yv;
        }
    }
}

// Fused tiny dt projection (K=4) + softplus for the per-head mamba.
__global__ void head_dt(const float* __restrict__ dbch, const float* __restrict__ dtw,
                        const float* __restrict__ dtb, float* __restrict__ o,
                        long long rows, int nch, int ld_dbc, int dtr)
{
    long long idx = (long long)blockIdx.x * blockDim.x + threadIdx.x;
    if (idx >= rows * nch) return;
    long long r = idx / nch;
    int c = (int)(idx % nch);
    float acc = dtb[c];
    for (int k = 0; k < dtr; ++k)
        acc += dbch[r * ld_dbc + k] * dtw[c * dtr + k];
    o[idx] = (acc > 20.f) ? acc : log1pf(expf(acc));
}

// xh[((h*B + b)*L + l)*dh + c] = x[(b*L+l)*d + c*nh + h]   (bf16 out)
__global__ void to_heads(const float* __restrict__ x, __bf16* __restrict__ xh,
                         int Bt, int L, int d, int nh, long long total)
{
    long long idx = (long long)blockIdx.x * blockDim.x + threadIdx.x;
    if (idx >= total) return;
    int dh = d / nh;
    int c = (int)(idx % dh);
    long long t = idx / dh;
    int l = (int)(t % L); t /= L;
    int b = (int)(t % Bt);
    int hh = (int)(t / Bt);
    xh[idx] = (__bf16)x[((long long)b * L + l) * d + c * nh + hh];
}

// x[(b*L+l)*d + c*nh + h] = yh[((h*B + b)*L + l)*dh + c]
__global__ void from_heads(const float* __restrict__ yh, float* __restrict__ x,
                           int Bt, int L, int d, int nh, long long total)
{
    long long idx = (long long)blockIdx.x * blockDim.x + threadIdx.x;
    if (idx >= total) return;
    int dh = d / nh;
    int c = (int)(idx % dh);
    long long t = idx / dh;
    int l = (int)(t % L); t /= L;
    int b = (int)(t % Bt);
    int hh = (int)(t / Bt);
    x[((long long)b * L + l) * d + c * nh + hh] = yh[idx];
}

// ---------------------------------------------------------------------------
// Host side
// ---------------------------------------------------------------------------
static inline int cdiv_ll(long long a, int b) { return (int)((a + b - 1) / b); }

extern "C" void kernel_launch(void* const* d_in, const int* in_sizes, int n_in,
                              void* d_out, int out_size, void* d_ws, size_t ws_size,
                              hipStream_t stream)
{
    (void)in_sizes; (void)n_in; (void)out_size; (void)ws_size;

    const int B = 4, L = 1024, D = 512, NH = 8;
    const int DI = 1024, DS = 16, DTR = 32;
    const int DH = 64, DIH = 128, DTRH = 4, BH = B * NH;      // 32
    const long long ROWS  = (long long)B * L;                  // 4096
    const long long ROWSH = (long long)BH * L;                 // 32768

    const float* x       = (const float*)d_in[0];
    const float* ln1_g   = (const float*)d_in[1];
    const float* ln1_b   = (const float*)d_in[2];
    const float* ln2_g   = (const float*)d_in[3];
    const float* ln2_b   = (const float*)d_in[4];
    const float* conv0_w = (const float*)d_in[5];
    const float* conv0_b = (const float*)d_in[6];
    const float* conv1_w = (const float*)d_in[7];
    const float* conv1_b = (const float*)d_in[8];
    const float* m_in_w  = (const float*)d_in[9];
    const float* m_cv_w  = (const float*)d_in[10];
    const float* m_cv_b  = (const float*)d_in[11];
    const float* m_xp_w  = (const float*)d_in[12];
    const float* m_dt_w  = (const float*)d_in[13];
    const float* m_dt_b  = (const float*)d_in[14];
    const float* m_Alog  = (const float*)d_in[15];
    const float* m_D     = (const float*)d_in[16];
    const float* m_out_w = (const float*)d_in[17];
    const float* h_in_w  = (const float*)d_in[18];
    const float* h_cv_w  = (const float*)d_in[19];
    const float* h_cv_b  = (const float*)d_in[20];
    const float* h_xp_w  = (const float*)d_in[21];
    const float* h_dt_w  = (const float*)d_in[22];
    const float* h_dt_b  = (const float*)d_in[23];
    const float* h_Alog  = (const float*)d_in[24];
    const float* h_D     = (const float*)d_in[25];
    const float* h_out_w = (const float*)d_in[26];
    float* out = (float*)d_out;

    char* wp = (char*)d_ws;
    auto alloc = [&](size_t bytes) -> void* {
        void* r = (void*)wp;
        wp += (bytes + 255) & ~(size_t)255;
        return r;
    };

    // Persistent bf16 weight packs
    __bf16* W0p    = (__bf16*)alloc(3LL * L * L * 2);
    __bf16* W1p    = (__bf16*)alloc(3LL * L * L * 2);
    __bf16* inwb   = (__bf16*)alloc(2LL * DI * D * 2);        // (2048,512)
    __bf16* xprojb = (__bf16*)alloc((long long)(DTR + 2 * DS) * DI * 2); // (64,1024)
    __bf16* dtwb   = (__bf16*)alloc((long long)DI * DTR * 2); // (1024,32)
    __bf16* outwb  = (__bf16*)alloc((long long)D * DI * 2);   // (512,1024)
    __bf16* inwhb  = (__bf16*)alloc(2LL * DIH * DH * 2);      // (256,64)
    __bf16* xprojhb= (__bf16*)alloc(48LL * DIH * 2);          // padded (48,128)
    __bf16* outwhb = (__bf16*)alloc((long long)DH * DIH * 2); // (64,128)
    float*  Aneg   = (float*)alloc((long long)DI * DS * 4);
    float*  Anegh  = (float*)alloc((long long)DIH * DS * 4);

    // Activations (aliased across phases; lifetimes are disjoint)
    float*  F0      = (float*)alloc(ROWS * D * 4);          // t0, later t3
    float*  convraw = (float*)alloc(ROWS * D * 4);          // conv0/conv1 GEMM out
    float*  F2      = (float*)alloc(ROWS * D * 4);          // t1, later t4
    float*  F3      = (float*)alloc(ROWS * D * 4);          // t2, later yh (8MB each)
    __bf16* t1b     = (__bf16*)alloc(ROWS * D * 2);         // later xh bf16 (same 4MB)
    __bf16* Xt      = (__bf16*)alloc((long long)B * (D + 2) * L * 2);
    float*  xzbuf   = (float*)alloc(ROWS * 2 * DI * 4);     // xz, later xzh (32MB)
    float*  u1      = (float*)alloc(ROWS * DI * 4);         // later u1h (16MB)
    __bf16* u1b     = (__bf16*)alloc(ROWS * DI * 2);        // later u1hb (8MB)
    float*  dtbuf   = (float*)alloc(ROWS * DI * 4);         // dtlin/dt, later dth (16MB)
    float*  dbc     = (float*)alloc(ROWS * 64 * 4);
    __bf16* dbcb    = (__bf16*)alloc(ROWS * 64 * 2);
    float*  dbch    = (float*)alloc(ROWSH * 48 * 4);
    __bf16* ybf     = (__bf16*)alloc(ROWSH * DIH * 2);      // yb / ybh (8MB)

    const int TB = 256;
    auto gemm = [&](const __bf16* A, const __bf16* Bt, float* C, int M, int N, int K,
                    int lda, int ldbt, int ldc, long long sA, long long sB, long long sC,
                    int batch, int accum) {
        if ((M % 64 == 0) && (N % 64 == 0)) {
            gemm_bf16_wmma_blk<<<dim3(N / 64, M / 64, batch), 128, 0, stream>>>(
                A, Bt, C, K, lda, ldbt, ldc, sA, sB, sC, accum);
        } else {
            gemm_bf16_wmma<<<dim3(N / 16, M / 16, batch), 32, 0, stream>>>(
                A, Bt, C, K, lda, ldbt, ldc, sA, sB, sC, accum);
        }
    };

    // ---- weight packing (every call; deterministic) ----
    pack_convw<<<cdiv_ll(3LL * L * L, TB), TB, 0, stream>>>(conv0_w, W0p, L);
    pack_convw<<<cdiv_ll(3LL * L * L, TB), TB, 0, stream>>>(conv1_w, W1p, L);
    cvt_bf16<<<cdiv_ll(2LL * DI * D, TB), TB, 0, stream>>>(m_in_w, inwb, 2LL * DI * D);
    cvt_bf16<<<cdiv_ll(64LL * DI, TB), TB, 0, stream>>>(m_xp_w, xprojb, 64LL * DI);
    cvt_bf16<<<cdiv_ll((long long)DI * DTR, TB), TB, 0, stream>>>(m_dt_w, dtwb, (long long)DI * DTR);
    cvt_bf16<<<cdiv_ll((long long)D * DI, TB), TB, 0, stream>>>(m_out_w, outwb, (long long)D * DI);
    cvt_bf16<<<cdiv_ll(2LL * DIH * DH, TB), TB, 0, stream>>>(h_in_w, inwhb, 2LL * DIH * DH);
    pack_pad_bf16<<<cdiv_ll(48LL * DIH, TB), TB, 0, stream>>>(h_xp_w, xprojhb, DTRH + 2 * DS, DIH, 48);
    cvt_bf16<<<cdiv_ll((long long)DH * DIH, TB), TB, 0, stream>>>(h_out_w, outwhb, (long long)DH * DIH);
    negexp<<<cdiv_ll(DI * DS, TB), TB, 0, stream>>>(m_Alog, Aneg, DI * DS);
    negexp<<<cdiv_ll(DIH * DS, TB), TB, 0, stream>>>(h_Alog, Anegh, DIH * DS);

    // ---- stage 1: LN1 -> dense seq conv0 + GeLU ----
    layernorm_k<<<(int)ROWS, TB, 0, stream>>>(x, ln1_g, ln1_b, F0, D);
    transpose_pad_bf16<<<dim3(cdiv_ll((long long)(D + 2) * L, TB), B), TB, 0, stream>>>(F0, Xt, L, D);
    for (int tap = 0; tap < 3; ++tap)
        gemm(W0p + (long long)tap * L * L, Xt + tap * L, convraw,
             L, D, L, L, L, D, 0, (long long)(D + 2) * L, (long long)L * D, B, tap > 0);
    bias_gelu<<<cdiv_ll(ROWS * D, TB), TB, 0, stream>>>(convraw, conv0_b, F2, L, D, ROWS * D);
    cvt_bf16<<<cdiv_ll(ROWS * D, TB), TB, 0, stream>>>(F2, t1b, ROWS * D);

    // ---- stage 2: full-width mamba ----
    gemm(t1b, inwb, xzbuf, (int)ROWS, 2 * DI, D, D, D, 2 * DI, 0, 0, 0, 1, 0);
    dwconv_silu<<<cdiv_ll(ROWS * DI, TB), TB, 0, stream>>>(xzbuf, m_cv_w, m_cv_b, u1, u1b,
                                                           L, DI, 2 * DI, ROWS * DI);
    gemm(u1b, xprojb, dbc, (int)ROWS, 64, DI, DI, DI, 64, 0, 0, 0, 1, 0);
    cvt_bf16<<<cdiv_ll(ROWS * 64, TB), TB, 0, stream>>>(dbc, dbcb, ROWS * 64);
    gemm(dbcb, dtwb, dtbuf, (int)ROWS, DI, DTR, 64, DTR, DI, 0, 0, 0, 1, 0);
    softplus_bias<<<cdiv_ll(ROWS * DI, TB), TB, 0, stream>>>(dtbuf, m_dt_b, ROWS * DI, DI);
    mamba_scan<<<dim3(DI / 16, B), 256, 0, stream>>>(dtbuf, u1, dbc, 64, DTR, DTR + DS,
                                                     xzbuf + DI, 2 * DI, Aneg, m_D,
                                                     ybf, L, DI);
    gemm(ybf, outwb, F3, (int)ROWS, D, DI, DI, DI, D, 0, 0, 0, 1, 0);

    // ---- stage 3: per-head mamba ----
    __bf16* xhb = t1b;   // reuse (same 4MB)
    to_heads<<<cdiv_ll(ROWS * D, TB), TB, 0, stream>>>(F3, xhb, B, L, D, NH, ROWS * D);
    float*  xzh  = xzbuf;
    float*  u1h  = u1;
    __bf16* u1hb = u1b;
    float*  dth  = dtbuf;
    __bf16* ybh  = ybf;
    float*  yh   = F3;   // t2 dead after to_heads
    gemm(xhb, inwhb, xzh, (int)ROWSH, 2 * DIH, DH, DH, DH, 2 * DIH, 0, 0, 0, 1, 0);
    dwconv_silu<<<cdiv_ll(ROWSH * DIH, TB), TB, 0, stream>>>(xzh, h_cv_w, h_cv_b, u1h, u1hb,
                                                             L, DIH, 2 * DIH, ROWSH * DIH);
    gemm(u1hb, xprojhb, dbch, (int)ROWSH, 48, DIH, DIH, DIH, 48, 0, 0, 0, 1, 0);
    head_dt<<<cdiv_ll(ROWSH * DIH, TB), TB, 0, stream>>>(dbch, h_dt_w, h_dt_b, dth,
                                                         ROWSH, DIH, 48, DTRH);
    mamba_scan<<<dim3(DIH / 16, BH), 256, 0, stream>>>(dth, u1h, dbch, 48, DTRH, DTRH + DS,
                                                       xzh + DIH, 2 * DIH, Anegh, h_D,
                                                       ybh, L, DIH);
    gemm(ybh, outwhb, yh, (int)ROWSH, DH, DIH, DIH, DIH, DH, 0, 0, 0, 1, 0);

    // ---- stage 4: un-reshape -> conv1 + GeLU -> LN2 ----
    float* t3 = F0;
    from_heads<<<cdiv_ll(ROWS * D, TB), TB, 0, stream>>>(yh, t3, B, L, D, NH, ROWS * D);
    transpose_pad_bf16<<<dim3(cdiv_ll((long long)(D + 2) * L, TB), B), TB, 0, stream>>>(t3, Xt, L, D);
    for (int tap = 0; tap < 3; ++tap)
        gemm(W1p + (long long)tap * L * L, Xt + tap * L, convraw,
             L, D, L, L, L, D, 0, (long long)(D + 2) * L, (long long)L * D, B, tap > 0);
    float* t4 = F2;
    bias_gelu<<<cdiv_ll(ROWS * D, TB), TB, 0, stream>>>(convraw, conv1_b, t4, L, D, ROWS * D);
    layernorm_k<<<(int)ROWS, TB, 0, stream>>>(t4, ln2_g, ln2_b, out, D);
}
